// MoEResBlock_9560597201202
// MI455X (gfx1250) — matrix operand order repaired and checked by reading, not compiled, not run on hardware
//
#include <hip/hip_runtime.h>
#include <hip/hip_bf16.h>

typedef __attribute__((ext_vector_type(16))) __bf16 v16bf;
typedef __attribute__((ext_vector_type(8)))  __bf16 bf16x8;
typedef __attribute__((ext_vector_type(8)))  float  v8f;
typedef __attribute__((ext_vector_type(4)))  unsigned int u32x4;
typedef __attribute__((ext_vector_type(4)))  int i32x4;
typedef __attribute__((ext_vector_type(8)))  int i32x8;

#define NTOK  65536
#define DDIM  256
#define NEXP  8
#define CAP   16384
#define NSLOT (NTOK*2)
#define LN_EPS 1e-6f

// ---------------- router: logits -> top2 -> softmax gates -------------------
__global__ __launch_bounds__(256) void router_kernel(
    const float* __restrict__ x0, const float* __restrict__ Wr,
    const float* __restrict__ br, int* __restrict__ eidx, float* __restrict__ gate)
{
  __shared__ float sW[DDIM*NEXP];
  __shared__ float sB[NEXP];
  int tid = threadIdx.x;
  for (int i = tid; i < DDIM*NEXP; i += 256) sW[i] = Wr[i];
  if (tid < NEXP) sB[tid] = br[tid];
  __syncthreads();
  int tok = blockIdx.x*256 + tid;
  const float* xr = x0 + (size_t)tok*DDIM;
  float acc[NEXP];
  #pragma unroll
  for (int e=0;e<NEXP;e++) acc[e] = sB[e];
  for (int d=0; d<DDIM; d+=4) {
    float4 x = *(const float4*)(xr + d);
    #pragma unroll
    for (int e=0;e<NEXP;e++)
      acc[e] += x.x*sW[(d+0)*NEXP+e] + x.y*sW[(d+1)*NEXP+e]
              + x.z*sW[(d+2)*NEXP+e] + x.w*sW[(d+3)*NEXP+e];
  }
  float v0 = -3.4e38f, v1 = -3.4e38f; int i0=0, i1=0;
  #pragma unroll
  for (int e=0;e<NEXP;e++) {
    float v = acc[e];
    if (v > v0)      { v1=v0; i1=i0; v0=v; i0=e; }
    else if (v > v1) { v1=v;  i1=e; }
  }
  float e1 = __expf(v1 - v0);
  float den = 1.0f + e1;
  eidx[2*tok+0] = i0;        eidx[2*tok+1] = i1;
  gate[2*tok+0] = 1.0f/den;  gate[2*tok+1] = e1/den;
}

// ------- convert f32 weights -> bf16, PRE-SWIZZLED into WMMA B layout -------
__global__ void cvt_swz_kernel(const float* __restrict__ src, __bf16* __restrict__ dst) {
  int i = blockIdx.x*256 + threadIdx.x;
  int stride = gridDim.x*256;
  for (; i < NEXP*DDIM*DDIM; i += stride) {
    int e = i >> 16, idx = i & 65535;
    int k = idx >> 8, n = idx & 255;
    int t = n >> 4,  nl = n & 15;
    int kk = k >> 5, kr = k & 31;
    int half = kr >> 4, arr = kr & 15;
    int sw = ((((t*8)+kk)*32) + (half*16+nl))*16 + arr;
    dst[(size_t)e*65536 + sw] = (__bf16)src[i];
  }
}

__global__ void zero_kernel(int4* __restrict__ p, int n4) {
  int i = blockIdx.x*blockDim.x + threadIdx.x;
  int stride = gridDim.x*blockDim.x;
  int4 z; z.x = 0; z.y = 0; z.z = 0; z.w = 0;
  for (; i < n4; i += stride) p[i] = z;
}

// ---------------- exact capacity scan (matches cumsum ordering) -------------
__global__ __launch_bounds__(1024) void scan_kernel(
    const int* __restrict__ eidx, float* __restrict__ gate, int* __restrict__ cmb)
{
  __shared__ int hist[1024][NEXP];
  int t = threadIdx.x;
  const int CH = NSLOT/1024;            // 128
  int base = t*CH;
  int cnt[NEXP];
  #pragma unroll
  for (int e=0;e<NEXP;e++) cnt[e]=0;
  for (int i=0;i<CH;i++) cnt[eidx[base+i]]++;
  #pragma unroll
  for (int e=0;e<NEXP;e++) hist[t][e] = cnt[e];
  __syncthreads();
  if (t < NEXP) {
    int run = 0;
    for (int j=0;j<1024;j++) { int c = hist[j][t]; hist[j][t] = run; run += c; }
  }
  __syncthreads();
  int run[NEXP];
  #pragma unroll
  for (int e=0;e<NEXP;e++) run[e] = hist[t][e];
  for (int i=0;i<CH;i++) {
    int e = eidx[base+i];
    int p = run[e]++;
    bool keep = p < CAP;
    int pc  = keep ? p : (CAP-1);
    int idx = e*CAP + pc;
    cmb[base+i] = keep ? idx : (idx | 0x80000000);
    if (!keep) gate[base+i] = 0.0f;
  }
}

// ---------------- dispatch: f32 token row -> bf16 xin[e][pos] ---------------
__global__ __launch_bounds__(256) void scatter_kernel(
    const float* __restrict__ x0, const int* __restrict__ cmb, __bf16* __restrict__ xin)
{
  int s    = blockIdx.x*8 + (threadIdx.x>>5);
  int lane = threadIdx.x & 31;
  int c = cmb[s];
  if (c < 0) return;
  const float* src = x0 + (size_t)(s>>1)*DDIM + lane*8;
  __bf16*      dst = xin + (size_t)c*DDIM + lane*8;
  float4 a = *(const float4*)(src);
  float4 b = *(const float4*)(src+4);
  bf16x8 o;
  o[0]=(__bf16)a.x; o[1]=(__bf16)a.y; o[2]=(__bf16)a.z; o[3]=(__bf16)a.w;
  o[4]=(__bf16)b.x; o[5]=(__bf16)b.y; o[6]=(__bf16)b.z; o[7]=(__bf16)b.w;
  *(bf16x8*)dst = o;
}

// ---------------- grouped GEMM + bias + LayerNorm (+ReLU), bf16 WMMA --------
// Weights pre-swizzled in global; one TDM descriptor DMAs the whole 128KB
// matrix into LDS (TENSORcnt). Fragment reads use two LDS base registers
// (second one made opaque via inline asm so LLVM cannot re-fold it into
// per-load VALU adds) -> every ds_load is base+imm16, no WMMA hazard NOPs.
template<bool RELU>
__global__ __launch_bounds__(256) void expert_gemm_ln_kernel(
    const __bf16* __restrict__ Abuf,   // [E*CAP, 256] bf16 row-major
    const __bf16* __restrict__ Wswz,   // [E, 65536] bf16, pre-swizzled B layout
    const float*  __restrict__ bias,
    const float*  __restrict__ lns,
    const float*  __restrict__ lnb,
    __bf16* __restrict__ Obuf)
{
  extern __shared__ __bf16 sW[];       // 65536 bf16 = 128KB
  const int e      = blockIdx.y;
  const int rowBlk = blockIdx.x * 128;

  // ---- TDM: flat 1-D tile copy, global (pre-swizzled) -> LDS ----
  if ((threadIdx.x >> 5) == 0) {       // wave 0 issues the DMA
    unsigned lds_off = (unsigned)(size_t)(&sW[0]);
    unsigned long long ga = (unsigned long long)(const void*)(Wswz + (size_t)e*65536);
    u32x4 g0v;
    g0v[0] = 1u;                                                  // count=1
    g0v[1] = lds_off;                                             // lds_addr
    g0v[2] = (unsigned)(ga & 0xffffffffu);                        // addr[31:0]
    g0v[3] = (unsigned)((ga >> 32) & 0x01ffffffu) | 0x80000000u;  // addr[56:32], type=2
    i32x8 g1v;
    g1v[0] = 0x20000;            // data_size=2 -> 4-byte elements
    g1v[1] = (int)0x80000000u;   // tensor_dim0 = 32768
    g1v[2] = 0x10000;            // tensor_dim1 = 1
    g1v[3] = (int)0x80000000u;   // tile_dim0   = 32768
    g1v[4] = 0;
    g1v[5] = 32768;              // tensor_dim0_stride
    g1v[6] = 0;
    g1v[7] = 0;
    i32x4 z4 = {0,0,0,0};
#if defined(__clang_major__) && (__clang_major__ >= 23)
    i32x8 z8 = {0,0,0,0,0,0,0,0};
    __builtin_amdgcn_tensor_load_to_lds(g0v, g1v, z4, z4, z8, 0);
#else
    __builtin_amdgcn_tensor_load_to_lds(g0v, g1v, z4, z4, 0);
#endif
    __builtin_amdgcn_s_wait_tensorcnt(0);
  }
  __syncthreads();

  const int lane = threadIdx.x & 31;
  const int wave = threadIdx.x >> 5;
  const int m = lane & 15, half = lane >> 4;
  const int rowBase = rowBlk + wave*16;
  const __bf16* arow = Abuf + ((size_t)e*CAP + rowBase + m)*DDIM;

  // two LDS base registers; hi displacement is opaque so it cannot be folded
  const char* bl0 = (const char*)sW + lane*32;
  unsigned hi_off = 65536;
  asm volatile("" : "+v"(hi_off));     // pin +64KB base into its own VGPR
  const char* bl1 = bl0 + hi_off;

  v8f acc[16];
  v8f vz = {0.f,0.f,0.f,0.f,0.f,0.f,0.f,0.f};
  #pragma unroll
  for (int t=0;t<16;t++) acc[t] = vz;

  #define LOAD_A(kk_) ({                                                      \
      bf16x8 alo_ = *(const bf16x8*)(arow + (kk_)*32 + half*16);              \
      bf16x8 ahi_ = *(const bf16x8*)(arow + (kk_)*32 + half*16 + 8);          \
      __builtin_shufflevector(alo_, ahi_,                                     \
          0,1,2,3,4,5,6,7,8,9,10,11,12,13,14,15); })
  #define LOAD_B(kk_, t_) ({                                                  \
      const char* bb_ = ((t_) < 8) ? bl0 : bl1;                               \
      const int off_ = ((t_) & 7)*8192 + (kk_)*1024;                          \
      bf16x8 b0_ = *(const bf16x8*)(bb_ + off_);                              \
      bf16x8 b1_ = *(const bf16x8*)(bb_ + off_ + 16);                         \
      __builtin_shufflevector(b0_, b1_,                                       \
          0,1,2,3,4,5,6,7,8,9,10,11,12,13,14,15); })

  v16bf aCur = LOAD_A(0);
  v16bf bCur = LOAD_B(0, 0);
  #pragma unroll
  for (int kk=0;kk<8;kk++) {
    v16bf aNext = aCur;
    if (kk < 7) aNext = LOAD_A(kk+1);        // 16 WMMAs of latency to cover
    #pragma unroll
    for (int t=0;t<16;t++) {
      v16bf bNext = bCur;
      if (t < 15)      bNext = LOAD_B(kk, t+1);
      else if (kk < 7) bNext = LOAD_B(kk+1, 0);
      acc[t] = __builtin_amdgcn_wmma_f32_16x16x32_bf16(
          false, aCur, false, bCur, (short)0, acc[t], false, false);
      bCur = bNext;
    }
    aCur = aNext;
  }
  #undef LOAD_A
  #undef LOAD_B

  // ---- epilogue: bias + LayerNorm over 256 cols (+optional ReLU) ----
  float bcol[16], scol[16], ocol[16];
  #pragma unroll
  for (int t=0;t<16;t++) {
    int col = t*16 + m;
    bcol[t] = bias[e*DDIM + col];
    scol[t] = lns [e*DDIM + col];
    ocol[t] = lnb [e*DDIM + col];
  }
  const float inv = 1.0f/(float)DDIM;
  #pragma unroll
  for (int r=0;r<8;r++) {
    float s=0.f, ss=0.f;
    #pragma unroll
    for (int t=0;t<16;t++) {
      float v = acc[t][r] + bcol[t];
      acc[t][r] = v;
      s += v; ss += v*v;
    }
    #pragma unroll
    for (int msk=1; msk<16; msk<<=1) {
      s  += __shfl_xor(s,  msk, 32);
      ss += __shfl_xor(ss, msk, 32);
    }
    float mu  = s*inv;
    float var = ss*inv - mu*mu;
    float rs  = rsqrtf(var + LN_EPS);
    size_t orow = ((size_t)e*CAP + rowBase + r + 8*half)*DDIM;
    #pragma unroll
    for (int t=0;t<16;t++) {
      float v = (acc[t][r]-mu)*rs*scol[t] + ocol[t];
      if (RELU) v = fmaxf(v, 0.f);
      Obuf[orow + t*16 + m] = (__bf16)v;
    }
  }
}

// ---------------- combine: gather 2 expert rows, gate, residual, relu -------
__global__ __launch_bounds__(256) void combine_kernel(
    const float* __restrict__ x0, const __bf16* __restrict__ ybuf,
    const int* __restrict__ cmb, const float* __restrict__ gate,
    float* __restrict__ out)
{
  int tok  = blockIdx.x*8 + (threadIdx.x>>5);
  int lane = threadIdx.x & 31;
  int s0 = 2*tok;
  int i0 = cmb[s0]   & 0x7fffffff;
  int i1 = cmb[s0+1] & 0x7fffffff;
  float g0 = gate[s0], g1 = gate[s0+1];
  int c = lane*8;
  bf16x8 y0 = *(const bf16x8*)(ybuf + (size_t)i0*DDIM + c);
  bf16x8 y1 = *(const bf16x8*)(ybuf + (size_t)i1*DDIM + c);
  const float* xr = x0 + (size_t)tok*DDIM + c;
  float4 xa = *(const float4*)xr;
  float4 xb = *(const float4*)(xr+4);
  float r[8] = {xa.x,xa.y,xa.z,xa.w,xb.x,xb.y,xb.z,xb.w};
  #pragma unroll
  for (int j=0;j<8;j++)
    r[j] = fmaxf(r[j] + g0*(float)y0[j] + g1*(float)y1[j], 0.f);
  float* od = out + (size_t)tok*DDIM + c;
  *(float4*)(od)   = make_float4(r[0],r[1],r[2],r[3]);
  *(float4*)(od+4) = make_float4(r[4],r[5],r[6],r[7]);
}

// ---------------- driver ----------------------------------------------------
extern "C" void kernel_launch(void* const* d_in, const int* in_sizes, int n_in,
                              void* d_out, int out_size, void* d_ws, size_t ws_size,
                              hipStream_t stream)
{
  const float* x0   = (const float*)d_in[0];
  const float* Wr   = (const float*)d_in[1];
  const float* br   = (const float*)d_in[2];
  const float* W1   = (const float*)d_in[3];
  const float* b1   = (const float*)d_in[4];
  const float* ln1s = (const float*)d_in[5];
  const float* ln1b = (const float*)d_in[6];
  const float* W2   = (const float*)d_in[7];
  const float* b2   = (const float*)d_in[8];
  const float* ln2s = (const float*)d_in[9];
  const float* ln2b = (const float*)d_in[10];
  float* out = (float*)d_out;

  char* ws = (char*)d_ws;
  size_t off = 0;
  int*    eidx = (int*)  (ws + off); off += (size_t)NSLOT*4;
  float*  gate = (float*)(ws + off); off += (size_t)NSLOT*4;
  int*    cmb  = (int*)  (ws + off); off += (size_t)NSLOT*4;
  __bf16* w1b  = (__bf16*)(ws + off); off += (size_t)NEXP*DDIM*DDIM*2;
  __bf16* w2b  = (__bf16*)(ws + off); off += (size_t)NEXP*DDIM*DDIM*2;
  off = (off + 255) & ~(size_t)255;
  __bf16* xin  = (__bf16*)(ws + off); off += (size_t)NEXP*CAP*DDIM*2;
  __bf16* hbuf = (__bf16*)(ws + off); off += (size_t)NEXP*CAP*DDIM*2;
  __bf16* ybuf = xin;   // xin dead after GEMM1 -> reuse as Y

  router_kernel<<<NTOK/256, 256, 0, stream>>>(x0, Wr, br, eidx, gate);
  cvt_swz_kernel<<<512, 256, 0, stream>>>(W1, w1b);
  cvt_swz_kernel<<<512, 256, 0, stream>>>(W2, w2b);
  zero_kernel<<<2048, 256, 0, stream>>>((int4*)xin,
      (int)(((size_t)NEXP*CAP*DDIM*2)/16));
  scan_kernel<<<1, 1024, 0, stream>>>(eidx, gate, cmb);
  scatter_kernel<<<NSLOT/8, 256, 0, stream>>>(x0, cmb, xin);

  dim3 g(CAP/128, NEXP);
  expert_gemm_ln_kernel<true ><<<g, 256, 131072, stream>>>(xin,  w1b, b1, ln1s, ln1b, hbuf);
  expert_gemm_ln_kernel<false><<<g, 256, 131072, stream>>>(hbuf, w2b, b2, ln2s, ln2b, ybuf);

  combine_kernel<<<NTOK/8, 256, 0, stream>>>(x0, ybuf, cmb, gate, out);
}
